// GraphBiasAttention_77472620085311
// MI455X (gfx1250) — matrix operand order repaired and checked by reading, not compiled, not run on hardware
//
#include <hip/hip_runtime.h>
#include <stdint.h>

typedef __bf16 bf16;
typedef __attribute__((ext_vector_type(16))) __bf16 v16bf;
typedef __attribute__((ext_vector_type(8)))  float  v8f;

#define BATCH   16
#define SEQ     1024
#define DMODEL  512
#define NHEADS  8
#define DK      64

union FragBF { uint4 u[2]; v16bf v; bf16 e[16]; };

static __device__ inline v8f wmma_bf16(v16bf a, v16bf b, v8f c) {
  return __builtin_amdgcn_wmma_f32_16x16x32_bf16(false, a, false, b,
                                                 (short)0, c, false, false);
}

// A-operand fragment (16x32 bf16): lane<16 holds K0..15, lane>=16 holds K16..31 (contiguous)
static __device__ inline v16bf load_a_frag(const bf16* row, int base8) {
  FragBF f;
  f.u[0] = *reinterpret_cast<const uint4*>(row + base8 * 2);
  f.u[1] = *reinterpret_cast<const uint4*>(row + base8 * 2 + 8);
  return f.v;
}
// B-operand fragment (32x16 bf16): lane<16 holds K0..15, lane>=16 holds K16..31 (contiguous)
static __device__ inline v16bf load_b_frag(const bf16* row, int base16) {
  FragBF f;
  f.u[0] = *reinterpret_cast<const uint4*>(row + base16);
  f.u[1] = *reinterpret_cast<const uint4*>(row + base16 + 8);
  return f.v;
}

// ---------------- f32 -> bf16 conversion ----------------
__global__ void cvt_f32_bf16(const float* __restrict__ in, bf16* __restrict__ out, int n) {
  int i = blockIdx.x * blockDim.x + threadIdx.x;
  int stride = gridDim.x * blockDim.x;
  for (; i < n; i += stride) out[i] = (bf16)in[i];
}

// ---------------- GEMM: out = A(MxK) * Wt(NxK)^T + bias ----------------
// Block = 8 waves; wave w computes a 32(m) x 64(n) tile: 2 A-frags reused
// against 4 B-frags -> 8 WMMAs per K-step (~22 FLOP/byte of L2 traffic).
template <bool OUT_F32>
__launch_bounds__(256)
__global__ void gemm_xwT(const bf16* __restrict__ A,
                         const bf16* __restrict__ Wt,
                         const float* __restrict__ bias,
                         void* __restrict__ outp,
                         int M, int N, int K) {
  const int lane  = threadIdx.x & 31;
  const int wave  = threadIdx.x >> 5;
  const int m0    = blockIdx.x * 32;
  const int n0    = wave * 64;
  const int l15   = lane & 15;
  const int base8 = (lane >= 16) ? 8 : 0;
  const int base16= (lane >= 16) ? 16 : 0;
  const int mvb   = (lane >= 16) ? 8 : 0;

  v8f acc[2][4] = {};
  const bf16* arow0 = A + (size_t)(m0 + l15) * K;
  const bf16* arow1 = A + (size_t)(m0 + 16 + l15) * K;
  const bf16* brow0 = Wt + (size_t)(n0 +  0 + l15) * K;
  const bf16* brow1 = Wt + (size_t)(n0 + 16 + l15) * K;
  const bf16* brow2 = Wt + (size_t)(n0 + 32 + l15) * K;
  const bf16* brow3 = Wt + (size_t)(n0 + 48 + l15) * K;

  for (int k = 0; k < K; k += 32) {
    v16bf a0 = load_a_frag(arow0 + k, base8);
    v16bf a1 = load_a_frag(arow1 + k, base8);
    v16bf b0 = load_b_frag(brow0 + k, base16);
    v16bf b1 = load_b_frag(brow1 + k, base16);
    v16bf b2 = load_b_frag(brow2 + k, base16);
    v16bf b3 = load_b_frag(brow3 + k, base16);
    acc[0][0] = wmma_bf16(a0, b0, acc[0][0]);
    acc[1][0] = wmma_bf16(a1, b0, acc[1][0]);
    acc[0][1] = wmma_bf16(a0, b1, acc[0][1]);
    acc[1][1] = wmma_bf16(a1, b1, acc[1][1]);
    acc[0][2] = wmma_bf16(a0, b2, acc[0][2]);
    acc[1][2] = wmma_bf16(a1, b2, acc[1][2]);
    acc[0][3] = wmma_bf16(a0, b3, acc[0][3]);
    acc[1][3] = wmma_bf16(a1, b3, acc[1][3]);
  }

#pragma unroll
  for (int r = 0; r < 2; ++r) {
#pragma unroll
    for (int t = 0; t < 4; ++t) {
      int n = n0 + t * 16 + l15;
      float bv = bias[n];
#pragma unroll
      for (int v = 0; v < 8; ++v) {
        float val = acc[r][t][v] + bv;
        size_t idx = (size_t)(m0 + r * 16 + mvb + v) * N + n;
        if (OUT_F32) ((float*)outp)[idx] = val;
        else         ((bf16*) outp)[idx] = (bf16)val;
      }
    }
  }
}

// ---------------- fused biased attention ----------------
// grid = (SEQ/16, NHEADS, BATCH), block = 256 (8 waves)
__launch_bounds__(256)
__global__ void attn_kernel(const bf16* __restrict__ Qb,
                            const bf16* __restrict__ Kb,
                            const bf16* __restrict__ Vb,
                            const float* __restrict__ bias,          // B x N x N
                            const unsigned char* __restrict__ mask,  // B x N
                            bf16* __restrict__ Ob) {                 // B x N x DMODEL
  __shared__ __align__(16) bf16  Pl[16][SEQ];      // 32 KB normalized probabilities
  __shared__ __align__(16) bf16  Vt[2][2][DK][40]; // 20 KB ping-pong transposed V staging
  __shared__ float Oacc[16 * DK];                  // 4 KB
  __shared__ float red[8][16];
  __shared__ float rowstat[16];

  const int tid   = threadIdx.x;
  const int lane  = tid & 31;
  const int wave  = tid >> 5;
  const int m0    = blockIdx.x * 16;
  const int h     = blockIdx.y;
  const int b     = blockIdx.z;
  const int l15   = lane & 15;
  const int base8 = (lane >= 16) ? 8 : 0;
  const int base16= (lane >= 16) ? 16 : 0;
  const int mvb   = (lane >= 16) ? 8 : 0;

  for (int i = tid; i < 16 * DK; i += 256) Oacc[i] = 0.f;

  // ---- Phase 1: S = Q K^T / 8 + bias (wave owns 128 columns) ----
  const bf16* qrow = Qb + (size_t)(b * SEQ + m0 + l15) * DMODEL + h * DK;
  v16bf qa0 = load_a_frag(qrow, base8);
  v16bf qa1 = load_a_frag(qrow + 32, base8);

  const int ncol0 = wave * 128;
  v8f s[8] = {};
#pragma unroll
  for (int t = 0; t < 8; ++t) {
    const bf16* krow = Kb + (size_t)(b * SEQ + ncol0 + t * 16 + l15) * DMODEL + h * DK;
    s[t] = wmma_bf16(qa0, load_b_frag(krow, base16), s[t]);
    s[t] = wmma_bf16(qa1, load_b_frag(krow + 32, base16), s[t]);
  }
#pragma unroll
  for (int t = 0; t < 8; ++t) {
    int n = ncol0 + t * 16 + l15;
    bool mk = mask[b * SEQ + n] != 0;
    const float* bp = bias + ((size_t)b * SEQ + (m0 + mvb)) * SEQ + n;
#pragma unroll
    for (int v = 0; v < 8; ++v) {
      float val = s[t][v] * 0.125f + bp[(size_t)v * SEQ];
      s[t][v] = mk ? val : -1e9f;
    }
  }

  // ---- Phase 2: softmax across 1024 columns (8 waves) ----
  float pmax[8];
#pragma unroll
  for (int v = 0; v < 8; ++v) {
    float x = s[0][v];
#pragma unroll
    for (int t = 1; t < 8; ++t) x = fmaxf(x, s[t][v]);
    for (int off = 1; off < 16; off <<= 1) x = fmaxf(x, __shfl_xor(x, off, 32));
    pmax[v] = x;
  }
  if (lane == 0 || lane == 16)
    for (int v = 0; v < 8; ++v) red[wave][mvb + v] = pmax[v];
  __syncthreads();
  if (tid < 16) {
    float x = red[0][tid];
    for (int w = 1; w < 8; ++w) x = fmaxf(x, red[w][tid]);
    rowstat[tid] = x;
  }
  __syncthreads();

  float rmax[8];
#pragma unroll
  for (int v = 0; v < 8; ++v) rmax[v] = rowstat[mvb + v];
  float psum[8];
#pragma unroll
  for (int v = 0; v < 8; ++v) {
    float acc = 0.f;
#pragma unroll
    for (int t = 0; t < 8; ++t) {
      float e = __expf(s[t][v] - rmax[v]);
      s[t][v] = e;
      acc += e;
    }
    for (int off = 1; off < 16; off <<= 1) acc += __shfl_xor(acc, off, 32);
    psum[v] = acc;
  }
  if (lane == 0 || lane == 16)
    for (int v = 0; v < 8; ++v) red[wave][mvb + v] = psum[v];
  __syncthreads();
  if (tid < 16) {
    float x = 0.f;
    for (int w = 0; w < 8; ++w) x += red[w][tid];
    rowstat[tid] = 1.0f / x;
  }
  __syncthreads();

#pragma unroll
  for (int v = 0; v < 8; ++v) {
    float inv = rowstat[mvb + v];
    int m = mvb + v;
#pragma unroll
    for (int t = 0; t < 8; ++t)
      Pl[m][ncol0 + t * 16 + l15] = (bf16)(s[t][v] * inv);
  }

  // ---- Phase 3: O = P V, waves 0-3 take K[0,512), waves 4-7 K[512,1024) ----
  // Ping-pong staged, transposed V tiles so staging overlaps WMMA.
  const int dt   = wave & 3;              // d-tile 0..3
  const int kgrp = (wave >> 2) * 512;
  const int khalf= wave >> 2;
  v8f oacc = {};

  const int sbuf = tid >> 7;              // staging role: which half of K
  const int st   = tid & 127;
  const int srow = st >> 2;               // 0..31
  const int scol = (st & 3) * 16;         // 0,16,32,48
  const bf16* vbase = Vb + (size_t)(b * SEQ + sbuf * 512 + srow) * DMODEL + h * DK + scol;

  // stage step 0 into buffer 0
  {
    FragBF c;
    c.u[0] = *reinterpret_cast<const uint4*>(vbase);
    c.u[1] = *reinterpret_cast<const uint4*>(vbase + 8);
#pragma unroll
    for (int j = 0; j < 16; ++j) Vt[0][sbuf][scol + j][srow] = c.e[j];
  }
  __syncthreads();   // also covers Pl writes from phase 2

  for (int ks = 0; ks < 16; ++ks) {
    // stage step ks+1 into the other buffer (overlaps with compute below)
    if (ks + 1 < 16) {
      const bf16* vrow = vbase + (size_t)(ks + 1) * 32 * DMODEL;
      FragBF c;
      c.u[0] = *reinterpret_cast<const uint4*>(vrow);
      c.u[1] = *reinterpret_cast<const uint4*>(vrow + 8);
#pragma unroll
      for (int j = 0; j < 16; ++j) Vt[(ks + 1) & 1][sbuf][scol + j][srow] = c.e[j];
    }

    int kk = kgrp + ks * 32;
    FragBF pa;
    pa.u[0] = *reinterpret_cast<const uint4*>(&Pl[l15][kk + base16]);
    pa.u[1] = *reinterpret_cast<const uint4*>(&Pl[l15][kk + base16 + 8]);
    int d = dt * 16 + l15;
    FragBF vf;
    vf.u[0] = *reinterpret_cast<const uint4*>(&Vt[ks & 1][khalf][d][base16]);
    vf.u[1] = *reinterpret_cast<const uint4*>(&Vt[ks & 1][khalf][d][base16 + 8]);
    oacc = wmma_bf16(pa.v, vf.v, oacc);
    __syncthreads();
  }

#pragma unroll
  for (int v = 0; v < 8; ++v)
    atomicAdd(&Oacc[(mvb + v) * DK + dt * 16 + l15], oacc[v]);
  __syncthreads();

  for (int i = tid; i < 16 * DK; i += 256) {
    int m = i >> 6, d = i & 63;
    Ob[(size_t)(b * SEQ + m0 + m) * DMODEL + h * DK + d] = (bf16)Oacc[i];
  }
}

extern "C" void kernel_launch(void* const* d_in, const int* in_sizes, int n_in,
                              void* d_out, int out_size, void* d_ws, size_t ws_size,
                              hipStream_t stream) {
  const float* x    = (const float*)d_in[0];
  const float* bias = (const float*)d_in[1];
  const unsigned char* mask = (const unsigned char*)d_in[2];
  const float* Wq = (const float*)d_in[3];
  const float* bq = (const float*)d_in[4];
  const float* Wk = (const float*)d_in[5];
  const float* bk = (const float*)d_in[6];
  const float* Wv = (const float*)d_in[7];
  const float* bv = (const float*)d_in[8];
  const float* Wo = (const float*)d_in[9];
  const float* bo = (const float*)d_in[10];

  const size_t XN = (size_t)BATCH * SEQ * DMODEL;   // 8,388,608
  const size_t WN = (size_t)DMODEL * DMODEL;        // 262,144

  bf16* p   = (bf16*)d_ws;
  bf16* xb  = p; p += XN;
  bf16* wqb = p; p += WN;
  bf16* wkb = p; p += WN;
  bf16* wvb = p; p += WN;
  bf16* wob = p; p += WN;
  bf16* Qb  = p; p += XN;
  bf16* Kb  = p; p += XN;
  bf16* Vb  = p; p += XN;
  bf16* Ab  = p; p += XN;   // attention output (bf16)

  cvt_f32_bf16<<<4096, 256, 0, stream>>>(x,  xb,  (int)XN);
  cvt_f32_bf16<<<512,  256, 0, stream>>>(Wq, wqb, (int)WN);
  cvt_f32_bf16<<<512,  256, 0, stream>>>(Wk, wkb, (int)WN);
  cvt_f32_bf16<<<512,  256, 0, stream>>>(Wv, wvb, (int)WN);
  cvt_f32_bf16<<<512,  256, 0, stream>>>(Wo, wob, (int)WN);

  dim3 gg(BATCH * SEQ / 32);   // 512 blocks, 32 rows each
  gemm_xwT<false><<<gg, 256, 0, stream>>>(xb, wqb, bq, Qb, BATCH * SEQ, DMODEL, DMODEL);
  gemm_xwT<false><<<gg, 256, 0, stream>>>(xb, wkb, bk, Kb, BATCH * SEQ, DMODEL, DMODEL);
  gemm_xwT<false><<<gg, 256, 0, stream>>>(xb, wvb, bv, Vb, BATCH * SEQ, DMODEL, DMODEL);

  dim3 ga(SEQ / 16, NHEADS, BATCH);   // 8192 blocks
  attn_kernel<<<ga, 256, 0, stream>>>(Qb, Kb, Vb, bias, mask, Ab);

  gemm_xwT<true><<<gg, 256, 0, stream>>>(Ab, wob, bo, d_out, BATCH * SEQ, DMODEL, DMODEL);
}
